// Encoder_300647710970
// MI455X (gfx1250) — compile-verified, matrix operand order checked by reading, not run on hardware
//
#include <hip/hip_runtime.h>
#include <hip/hip_bf16.h>
#include <stdint.h>

// ---------------------------------------------------------------------------
// GCN encoder for MI455X (gfx1250, wave32).
//   FC + 3x(GEMM -> gather/scatter -> bias/ReLU -> BN(train) -> pool)
// GEMMs via v_wmma_f32_16x16x32_bf16 (f32 accumulate). Edge phase uses
// L2-resident buffers (51.2MB each << 192MB L2) + global_atomic_add_f32.
// Round 2: fp32->bf16 conversion now uses the native __bf16 cast so the
// backend can emit hardware cvt ops instead of 3-VALU software RNE.
// ---------------------------------------------------------------------------

#define DIMC 128          // hidden dim == feature dim
#define LNUM 3            // layers
#define BN_EPS 1e-5f

typedef __attribute__((ext_vector_type(16))) __bf16 v16bf;
typedef __attribute__((ext_vector_type(8)))  float  v8f;

static __device__ __forceinline__ __bf16 f2bf(float f) {
  return static_cast<__bf16>(f);   // native RNE conversion on gfx1250
}

static __device__ __forceinline__ float atomic_add_f32(float* p, float v) {
  return __hip_atomic_fetch_add(p, v, __ATOMIC_RELAXED, __HIP_MEMORY_SCOPE_AGENT);
}

// ---- WMMA fragment loaders (CDNA5 ISA 7.12.2 layouts, wave32) --------------
// 16-bit A matrix 16x32: lanes 0-15 -> M=lane, lanes 16-31 -> M=lane-16 with
// K offset +8; VGPR v holds K pair {base, base+1}, base = 2v (v<4) else 16+2(v-4).
static __device__ __forceinline__ v16bf load_A_frag(const float* __restrict__ A,
                                                    int lda, int row0, int k0,
                                                    int lane, int nrows) {
  int half = lane >> 4;
  int m = row0 + (lane & 15);
  if (m >= nrows) m = nrows - 1;            // clamp (keeps EXEC all-ones)
  const float* ap = A + (size_t)m * lda + k0 + half * 8;
  v16bf a;
#pragma unroll
  for (int v = 0; v < 8; ++v) {
    int base = (v < 4) ? (2 * v) : (16 + 2 * (v - 4));
    a[2 * v]     = f2bf(ap[base]);
    a[2 * v + 1] = f2bf(ap[base + 1]);
  }
  return a;
}

// B matrix 32x16 (K x N): lane%16 -> N column, same K striping as A.
// B[k][n] = W[n][k]  (we compute Out = A * W^T, W row-major [n][k], 128 wide)
static __device__ __forceinline__ v16bf load_B_frag(const float* __restrict__ W,
                                                    int n0, int k0, int lane) {
  int half = lane >> 4;
  int n = n0 + (lane & 15);
  const float* bp = W + (size_t)n * DIMC + k0 + half * 8;
  v16bf b;
#pragma unroll
  for (int v = 0; v < 8; ++v) {
    int base = (v < 4) ? (2 * v) : (16 + 2 * (v - 4));
    b[2 * v]     = f2bf(bp[base]);
    b[2 * v + 1] = f2bf(bp[base + 1]);
  }
  return b;
}

// ---- GEMM: Out[nrows x 128] = A[nrows x K=128, row stride lda] * W^T -------
// 256 threads = 8 waves; wave w -> 16x16 tile at columns [16w, 16w+16);
// block -> 16-row band. A-band reused by all 8 waves via L0/L2.
__global__ void wmma_gemm_nt(const float* __restrict__ A, int lda,
                             const float* __restrict__ W,
                             float* __restrict__ Out, int nrows) {
  int lane = threadIdx.x & 31;
  int wave = threadIdx.x >> 5;
  int row0 = blockIdx.x * 16;
  int n0 = wave * 16;
  v8f acc = {0.f, 0.f, 0.f, 0.f, 0.f, 0.f, 0.f, 0.f};
#pragma unroll
  for (int kk = 0; kk < DIMC; kk += 32) {
    v16bf a = load_A_frag(A, lda, row0, kk, lane, nrows);
    v16bf b = load_B_frag(W, n0, kk, lane);
    acc = __builtin_amdgcn_wmma_f32_16x16x32_bf16(
        /*neg_a=*/false, a, /*neg_b=*/false, b,
        /*c_mod=*/(short)0, acc, /*reuse_a=*/false, /*reuse_b=*/false);
  }
  // C/D layout: VGPR j holds row M = j + 8*half, lane%16 = column
  int half = lane >> 4;
  int n = n0 + (lane & 15);
#pragma unroll
  for (int j = 0; j < 8; ++j) {
    int m = row0 + j + 8 * half;
    if (m < nrows) Out[(size_t)m * DIMC + n] = acc[j];
  }
}

// ---- misc elementwise kernels ---------------------------------------------
__global__ void zero_f32(float* __restrict__ p, size_t n) {
  size_t i = (size_t)blockIdx.x * blockDim.x + threadIdx.x;
  size_t stride = (size_t)gridDim.x * blockDim.x;
  for (; i < n; i += stride) p[i] = 0.0f;
}

// deg[col[e]] += ew[e]   (edge_index is int64: [0..E) rows, [E..2E) cols)
__global__ void deg_accum(const long long* __restrict__ ei,
                          const float* __restrict__ ew,
                          float* __restrict__ deg, int E) {
  int e = blockIdx.x * blockDim.x + threadIdx.x;
  if (e < E) atomic_add_f32(&deg[ei[(size_t)E + e]], ew[e]);
}

// deg += 1 (self loop), then dinv = rsqrt(deg) in place
__global__ void make_dinv(float* __restrict__ deg, int n) {
  int i = blockIdx.x * blockDim.x + threadIdx.x;
  if (i < n) {
    float d = deg[i] + 1.0f;
    deg[i] = (d > 0.0f) ? rsqrtf(d) : 0.0f;
  }
}

// C[n][:] = bias + dinv[n]^2 * B[n][:]   (bias + self-loop message)
__global__ void init_C(const float* __restrict__ B, const float* __restrict__ dinv,
                       const float* __restrict__ bias, float* __restrict__ C, int n) {
  int idx = blockIdx.x * blockDim.x + threadIdx.x;   // n*32 threads (float4 each)
  if (idx >= n * 32) return;
  int node = idx >> 5;
  int f4 = (idx & 31) * 4;
  float s = dinv[node];
  s *= s;
  const float4 v = *(const float4*)(B + (size_t)node * DIMC + f4);
  const float4 bb = *(const float4*)(bias + f4);
  float4 o;
  o.x = bb.x + s * v.x; o.y = bb.y + s * v.y;
  o.z = bb.z + s * v.z; o.w = bb.w + s * v.w;
  *(float4*)(C + (size_t)node * DIMC + f4) = o;
}

// one wave per edge: C[col] += dinv[row]*ew*dinv[col] * B[row]
// lane covers 4 features (float4 gather), 4 f32 atomics scatter.
__global__ void scatter_edges(const long long* __restrict__ ei,
                              const float* __restrict__ ew,
                              const float* __restrict__ dinv,
                              const float* __restrict__ B,
                              float* __restrict__ C, int E) {
  int lane = threadIdx.x & 31;
  int e = blockIdx.x * (blockDim.x >> 5) + (threadIdx.x >> 5);
  if (e >= E) return;
  long long r = ei[e];
  long long c = ei[(size_t)E + e];
  float nrm = dinv[r] * ew[e] * dinv[c];
  const float4 v = ((const float4*)(B + (size_t)r * DIMC))[lane];
  float* dst = C + (size_t)c * DIMC + lane * 4;
  atomic_add_f32(dst + 0, nrm * v.x);
  atomic_add_f32(dst + 1, nrm * v.y);
  atomic_add_f32(dst + 2, nrm * v.z);
  atomic_add_f32(dst + 3, nrm * v.w);
}

// in-place ReLU + per-feature sum / sumsq (block-reduced in registers,
// one atomic per block per feature). blockDim = 128 (feature = threadIdx.x)
__global__ void relu_stats(float* __restrict__ C, float* __restrict__ stats, int n) {
  int d = threadIdx.x;
  float s = 0.0f, sq = 0.0f;
  for (int node = blockIdx.x; node < n; node += gridDim.x) {
    float v = C[(size_t)node * DIMC + d];
    v = v > 0.0f ? v : 0.0f;
    C[(size_t)node * DIMC + d] = v;
    s += v;
    sq += v * v;
  }
  atomic_add_f32(&stats[d], s);
  atomic_add_f32(&stats[DIMC + d], sq);
}

// BN normalize (training stats) -> write xs slab + atomic pool accumulate
__global__ void bn_pool(const float* __restrict__ C, const float* __restrict__ stats,
                        const float* __restrict__ gamma, const float* __restrict__ beta,
                        const long long* __restrict__ batch,
                        float* __restrict__ out_pool, float* __restrict__ out_xs,
                        int n, int col_off) {
  size_t idx = (size_t)blockIdx.x * blockDim.x + threadIdx.x;
  if (idx >= (size_t)n * DIMC) return;
  int node = (int)(idx >> 7);
  int d = (int)(idx & (DIMC - 1));
  float inv_n = 1.0f / (float)n;
  float mean = stats[d] * inv_n;
  float var = stats[DIMC + d] * inv_n - mean * mean;
  float v = (C[idx] - mean) * rsqrtf(var + BN_EPS) * gamma[d] + beta[d];
  size_t colcat = (size_t)(LNUM * DIMC);
  out_xs[(size_t)node * colcat + col_off + d] = v;
  atomic_add_f32(&out_pool[(size_t)batch[node] * colcat + col_off + d], v);
}

// ---------------------------------------------------------------------------
extern "C" void kernel_launch(void* const* d_in, const int* in_sizes, int n_in,
                              void* d_out, int out_size, void* d_ws, size_t ws_size,
                              hipStream_t stream) {
  const float*     x     = (const float*)d_in[0];       // [N,128] f32
  const long long* ei    = (const long long*)d_in[1];   // [2,E] i64
  const float*     ew    = (const float*)d_in[2];       // [E] f32
  const long long* batch = (const long long*)d_in[3];   // [N] i64
  const float*     fc_w  = (const float*)d_in[4];       // [128,128]
  const float*     Wl    = (const float*)d_in[5];       // [3,128,128]
  const float*     bl    = (const float*)d_in[6];       // [3,128]
  const float*     gl    = (const float*)d_in[7];       // [3,128]
  const float*     btl   = (const float*)d_in[8];       // [3,128]

  const int N = in_sizes[0] / DIMC;
  const int E = in_sizes[2];
  const int G = out_size / (LNUM * DIMC) - N;           // pooled graph count

  // workspace layout (floats): dinv | R1 (H / C accumulator) | R2 (hl) | stats
  float* ws = (float*)d_ws;
  size_t off = 0;
  float* dinv = ws + off;  off += ((size_t)N + 255) & ~(size_t)255;
  float* R1   = ws + off;  off += (size_t)N * DIMC;
  float* R2   = ws + off;  off += (size_t)N * DIMC;
  float* stats = ws + off; off += 256;
  (void)ws_size; (void)n_in;

  float* out_pool = (float*)d_out;                        // [G, 384]
  float* out_xs   = out_pool + (size_t)G * (LNUM * DIMC); // [N, 384]

  const int TB = 256;

  // --- gcn_norm: degrees (edges + self loop), dinv = rsqrt ---
  zero_f32<<<dim3((N + TB - 1) / TB), dim3(TB), 0, stream>>>(dinv, (size_t)N);
  zero_f32<<<dim3((unsigned)(((size_t)G * LNUM * DIMC + TB - 1) / TB)), dim3(TB), 0, stream>>>(
      out_pool, (size_t)G * LNUM * DIMC);
  deg_accum<<<dim3((E + TB - 1) / TB), dim3(TB), 0, stream>>>(ei, ew, dinv, E);
  make_dinv<<<dim3((N + TB - 1) / TB), dim3(TB), 0, stream>>>(dinv, N);

  // --- FC: H = x @ fc_w^T  (WMMA bf16, f32 accumulate) ---
  const int gemm_blocks = (N + 15) / 16;
  wmma_gemm_nt<<<dim3(gemm_blocks), dim3(TB), 0, stream>>>(x, DIMC, fc_w, R1, N);

  for (int l = 0; l < LNUM; ++l) {
    // hl = h @ W[l]^T ; layer 0 reads H from R1, later layers read the
    // previous layer's normalized output straight out of the xs slab.
    const float* A   = (l == 0) ? R1 : (out_xs + (size_t)(l - 1) * DIMC);
    const int    lda = (l == 0) ? DIMC : LNUM * DIMC;
    wmma_gemm_nt<<<dim3(gemm_blocks), dim3(TB), 0, stream>>>(
        A, lda, Wl + (size_t)l * DIMC * DIMC, R2, N);

    // C = bias + self-loop term (R1 is dead after the GEMM above -> reuse)
    init_C<<<dim3((N * 32 + TB - 1) / TB), dim3(TB), 0, stream>>>(
        R2, dinv, bl + (size_t)l * DIMC, R1, N);

    // edge gather * norm -> scatter-add (L2-resident src/dst)
    scatter_edges<<<dim3((E + 7) / 8), dim3(TB), 0, stream>>>(
        ei, ew, dinv, R2, R1, E);

    // ReLU + BN statistics
    zero_f32<<<dim3(1), dim3(TB), 0, stream>>>(stats, (size_t)256);
    relu_stats<<<dim3(1024), dim3(DIMC), 0, stream>>>(R1, stats, N);

    // normalize -> xs slab + pooled segment-sum
    bn_pool<<<dim3((unsigned)(((size_t)N * DIMC + TB - 1) / TB)), dim3(TB), 0, stream>>>(
        R1, stats, gl + (size_t)l * DIMC, btl + (size_t)l * DIMC, batch,
        out_pool, out_xs, N, l * DIMC);
  }
}
